// Gpt2Block_48060684042259
// MI455X (gfx1250) — matrix operand order repaired and checked by reading, not compiled, not run on hardware
//
#include <hip/hip_runtime.h>

// GPT-2 block for gfx1250 (MI455X): all GEMMs via v_wmma_f32_16x16x32_f16.
// B=2, S=2048, E=1024, H=16, DH=64, I=4096.
// GEMM consumes B in transposed (N x K) layout so both A and B tiles stage as
// contiguous 16B copies and both WMMA fragments are contiguous ds_load_b128s.
// Tile staging uses GLOBAL_LOAD_ASYNC_TO_LDS_B128 (ASYNCcnt) when available.

typedef __attribute__((ext_vector_type(16))) _Float16 v16h;
typedef __attribute__((ext_vector_type(8)))  _Float16 v8h;
typedef __attribute__((ext_vector_type(8)))  float    v8f;

static constexpr int Bc  = 2;
static constexpr int Sc  = 2048;
static constexpr int Ec  = 1024;
static constexpr int Hc  = 16;
static constexpr int DHc = 64;
static constexpr int Ic  = 4 * Ec;
static constexpr int Mc  = Bc * Sc;          // 4096 rows

// ---- CDNA5 async global->LDS path (guarded; falls back to sync staging) ----
#if __has_builtin(__builtin_amdgcn_global_load_async_to_lds_b128) && \
    __has_builtin(__builtin_amdgcn_s_wait_asynccnt)
#define ASYNC_LDS 1
#endif

#ifdef ASYNC_LDS
typedef int b128_t __attribute__((vector_size(16)));
typedef __attribute__((address_space(1))) b128_t g_b128;
typedef __attribute__((address_space(3))) b128_t l_b128;

__device__ __forceinline__ void cp16_async(_Float16* ldst, const _Float16* gsrc) {
  __builtin_amdgcn_global_load_async_to_lds_b128(
      (g_b128*)(void*)gsrc, (l_b128*)(void*)ldst, 0, 0);
}
#endif

// ---------------------------------------------------------------- utilities

// Tiled transpose + f32->f16 convert: W (K x N) -> WT (N x K), both row-major.
__global__ __launch_bounds__(256) void k_cvt_t(const float* __restrict__ src,
                                               _Float16* __restrict__ dst,
                                               int K, int N) {
  __shared__ float t[32][33];
  const int n0 = blockIdx.x * 32, k0 = blockIdx.y * 32;
  for (int j = threadIdx.y; j < 32; j += 8)
    t[j][threadIdx.x] = src[(size_t)(k0 + j) * N + n0 + threadIdx.x];
  __syncthreads();
  for (int j = threadIdx.y; j < 32; j += 8)
    dst[(size_t)(n0 + j) * K + k0 + threadIdx.x] = (_Float16)t[threadIdx.x][j];
}

// LayerNorm: one block per row of E=1024, emits f16 for WMMA consumption.
__global__ __launch_bounds__(256) void k_ln(const float* __restrict__ x,
                                            const float* __restrict__ g,
                                            const float* __restrict__ bb,
                                            _Float16* __restrict__ out, int E) {
  __shared__ float red[256];
  const int tid = threadIdx.x;
  const size_t row = blockIdx.x;
  const float* xr = x + row * (size_t)E;

  float s = 0.f;
  for (int i = tid; i < E; i += 256) s += xr[i];
  red[tid] = s; __syncthreads();
  for (int st = 128; st > 0; st >>= 1) {
    if (tid < st) red[tid] += red[tid + st];
    __syncthreads();
  }
  const float mu = red[0] / (float)E;
  __syncthreads();

  float v = 0.f;
  for (int i = tid; i < E; i += 256) { float d = xr[i] - mu; v += d * d; }
  red[tid] = v; __syncthreads();
  for (int st = 128; st > 0; st >>= 1) {
    if (tid < st) red[tid] += red[tid + st];
    __syncthreads();
  }
  const float rs = rsqrtf(red[0] / (float)E + 1e-5f);

  _Float16* orow = out + row * (size_t)E;
  for (int i = tid; i < E; i += 256)
    orow[i] = (_Float16)((xr[i] - mu) * rs * g[i] + bb[i]);
}

// Split qkv (M x 3E, f32) into per-head f16 operands:
//   q16: (B,H,S,DH) row-major, pre-scaled by 1/sqrt(DH)   (A of scores GEMM)
//   k16: (B,H,S,DH) row-major                              (B^T of scores GEMM)
//   vT16:(B,H,DH,S) row-major                              (B^T of ctx GEMM)
__global__ __launch_bounds__(256) void k_pack_qkv(const float* __restrict__ qkv,
                                                  _Float16* __restrict__ q16,
                                                  _Float16* __restrict__ k16,
                                                  _Float16* __restrict__ vT16) {
  size_t i = (size_t)blockIdx.x * 256 + threadIdx.x;
  if (i >= (size_t)Mc * Ec) return;
  int row = (int)(i / Ec);
  int col = (int)(i % Ec);
  int b  = row / Sc, s = row % Sc;
  int h  = col / DHc, d = col % DHc;
  size_t base = (size_t)row * (3 * Ec);
  float qv = qkv[base + col] * 0.125f;           // 1/sqrt(64)
  float kv = qkv[base + Ec + col];
  float vv = qkv[base + 2 * Ec + col];
  size_t bh = (size_t)(b * Hc + h);
  q16 [(bh * Sc + s) * DHc + d] = (_Float16)qv;
  k16 [(bh * Sc + s) * DHc + d] = (_Float16)kv;
  vT16[(bh * DHc + d) * Sc + s] = (_Float16)vv;
}

// Causal masked softmax in place. One block per (b,h,q) row of length S.
__global__ __launch_bounds__(256) void k_softmax(float* __restrict__ attn,
                                                 const float* __restrict__ mask,
                                                 int S, int H) {
  __shared__ float red[256];
  const int tid = threadIdx.x;
  const size_t row = blockIdx.x;           // B*H*S rows
  const int q = (int)(row % S);
  const size_t bh = row / S;
  const int b = (int)(bh / H);
  float* p = attn + row * (size_t)S;
  const float* mrow = mask + (size_t)b * S;
  const int lim = q + 1;

  float mx = -3.0e38f;
  for (int k = tid; k < lim; k += 256)
    if (mrow[k] == 1.0f) mx = fmaxf(mx, p[k]);
  red[tid] = mx; __syncthreads();
  for (int st = 128; st > 0; st >>= 1) {
    if (tid < st) red[tid] = fmaxf(red[tid], red[tid + st]);
    __syncthreads();
  }
  mx = red[0]; __syncthreads();

  float sum = 0.f;
  for (int k = tid; k < S; k += 256) {
    float e = 0.f;
    if (k < lim && mrow[k] == 1.0f) e = __expf(p[k] - mx);
    p[k] = e; sum += e;
  }
  red[tid] = sum; __syncthreads();
  for (int st = 128; st > 0; st >>= 1) {
    if (tid < st) red[tid] += red[tid + st];
    __syncthreads();
  }
  const float inv = 1.0f / red[0];
  for (int k = tid; k < S; k += 256) p[k] *= inv;
}

// ---------------------------------------------------------------- WMMA GEMM

__device__ __forceinline__ float gelu_f(float x) {
  const float c = 0.7978845608028654f;
  float t = tanhf(c * (x + 0.044715f * x * x * x));
  return 0.5f * x * (1.0f + t);
}

// Stage 8 contiguous elements into LDS (f16 async/sync, or f32 downconvert).
__device__ __forceinline__ void stage8(_Float16* dst, const _Float16* src) {
#ifdef ASYNC_LDS
  cp16_async(dst, src);
#else
  *(v8h*)dst = *(const v8h*)src;
#endif
}
__device__ __forceinline__ void stage8(_Float16* dst, const float* src) {
  float4 a = *(const float4*)src;
  float4 b = *(const float4*)(src + 4);
  dst[0] = (_Float16)a.x; dst[1] = (_Float16)a.y;
  dst[2] = (_Float16)a.z; dst[3] = (_Float16)a.w;
  dst[4] = (_Float16)b.x; dst[5] = (_Float16)b.y;
  dst[6] = (_Float16)b.z; dst[7] = (_Float16)b.w;
}

struct GemmP {
  const void*      A;       // AT (M x K), row-major, lda
  const _Float16*  Bt;      // f16 B^T (N x K), row-major, ldb
  const float*     bias;    // [N] or null
  const float*     resid;   // f32 (M x N) stride ldc, or null
  float*           outF;    // f32 out (M x N) stride ldc, or null
  _Float16*        outH;    // f16 out (M x N) stride ldh, or null
  int M, N, K;
  int lda, ldb, ldc, ldh;
  int zdiv;                 // batch z split: zo = z/zdiv, zi = z%zdiv
  long long aZ0, aZ1, bZ0, bZ1, cZ0, cZ1, hZ0, hZ1;
  int doGelu;
};

// Block tile 64(M) x 128(N), K-step 32; 8 waves, each wave 32x32 via 2x2
// v_wmma_f32_16x16x32_f16 tiles. Both As and Bs are [row][K] so fragments are
// contiguous 16B LDS reads matching the ISA 7.12.2 16-bit A/B layouts.
template <typename AT>
__global__ __launch_bounds__(256) void k_gemm(GemmP p) {
  __shared__ __align__(16) _Float16 As[64][40];    // [m][k], 80B rows
  __shared__ __align__(16) _Float16 Bs[128][40];   // [n][k], 80B rows

  const int tid   = threadIdx.x;
  const int lane  = tid & 31;
  const int wave  = tid >> 5;
  const int waveM = wave >> 2;          // 0..1  (rows of 32)
  const int waveN = wave & 3;           // 0..3  (cols of 32)

  const int z  = blockIdx.z;
  const int zo = (p.zdiv > 1) ? (z / p.zdiv) : z;
  const int zi = (p.zdiv > 1) ? (z % p.zdiv) : 0;

  const AT*       Ag = (const AT*)p.A + (size_t)zo * p.aZ0 + (size_t)zi * p.aZ1;
  const _Float16* Bg = p.Bt          + (size_t)zo * p.bZ0 + (size_t)zi * p.bZ1;
  float*          Cg = p.outF ? p.outF + (size_t)zo * p.cZ0 + (size_t)zi * p.cZ1 : nullptr;
  _Float16*       Hg = p.outH ? p.outH + (size_t)zo * p.hZ0 + (size_t)zi * p.hZ1 : nullptr;

  const int blockM = blockIdx.y * 64;
  const int blockN = blockIdx.x * 128;

  const v8f zero8 = {0.f, 0.f, 0.f, 0.f, 0.f, 0.f, 0.f, 0.f};
  v8f acc[2][2];
  acc[0][0] = zero8; acc[0][1] = zero8; acc[1][0] = zero8; acc[1][1] = zero8;

  const int aRow = tid >> 2;            // 0..63
  const int aCol = (tid & 3) * 8;       // 0,8,16,24
  const int bRow = tid >> 1;            // 0..127
  const int bCol = (tid & 1) * 16;      // 0,16
  const bool bInRange = (blockN + bRow) < p.N;

  const int mSel = lane & 15;
  const int hi   = lane >> 4;

  for (int k0 = 0; k0 < p.K; k0 += 32) {
    // ---- stage A tile (64 x 32): one 16B copy per thread
    stage8(&As[aRow][aCol],
           Ag + (size_t)(blockM + aRow) * p.lda + (k0 + aCol));

    // ---- stage B^T tile (128 x 32): two 16B copies per thread
    if (bInRange) {
      const _Float16* src = Bg + (size_t)(blockN + bRow) * p.ldb + (k0 + bCol);
      stage8(&Bs[bRow][bCol],     src);
      stage8(&Bs[bRow][bCol + 8], src + 8);
    } else {
      v8h zh = {};
      *(v8h*)&Bs[bRow][bCol]     = zh;
      *(v8h*)&Bs[bRow][bCol + 8] = zh;
    }

#ifdef ASYNC_LDS
    __builtin_amdgcn_s_wait_asynccnt(0);
#else
    if (k0 + 32 < p.K) {   // prefetch next K tiles (global_prefetch_b8)
      __builtin_prefetch(Ag + (size_t)(blockM + aRow) * p.lda + (k0 + 32 + aCol), 0, 0);
      if (bInRange)
        __builtin_prefetch(Bg + (size_t)(blockN + bRow) * p.ldb + (k0 + 32 + bCol), 0, 0);
    }
#endif
    __syncthreads();

    // ---- build fragments (per ISA 7.12.2 16-bit layouts)
    v16h aF[2], bF[2];
#pragma unroll
    for (int tm = 0; tm < 2; ++tm) {
      const int m  = waveM * 32 + tm * 16 + mSel;
      const int kb = hi * 8;                       // lanes 16-31: K+8 block
      v8h lo  = *(const v8h*)&As[m][kb];           // K = kb .. kb+7
      v8h hi8 = *(const v8h*)&As[m][kb + 16];      // K = 16+kb .. 16+kb+7
#pragma unroll
      for (int j = 0; j < 8; ++j) { aF[tm][j] = lo[j]; aF[tm][8 + j] = hi8[j]; }
    }
#pragma unroll
    for (int tn = 0; tn < 2; ++tn) {
      const int n  = waveN * 32 + tn * 16 + mSel;
      const int kb = hi * 8;                       // lanes 16-31: K+8 block
      v8h lo  = *(const v8h*)&Bs[n][kb];
      v8h hi8 = *(const v8h*)&Bs[n][kb + 16];
#pragma unroll
      for (int j = 0; j < 8; ++j) { bF[tn][j] = lo[j]; bF[tn][8 + j] = hi8[j]; }
    }

#pragma unroll
    for (int tm = 0; tm < 2; ++tm)
#pragma unroll
      for (int tn = 0; tn < 2; ++tn)
        acc[tm][tn] = __builtin_amdgcn_wmma_f32_16x16x32_f16(
            false, aF[tm], false, bF[tn], (short)0, acc[tm][tn], false, false);

    __syncthreads();
  }

  // ---- epilogue: bias, gelu, residual, f32/f16 stores
#pragma unroll
  for (int tm = 0; tm < 2; ++tm) {
#pragma unroll
    for (int tn = 0; tn < 2; ++tn) {
      const int gn = blockN + waveN * 32 + tn * 16 + mSel;
      if (gn >= p.N) continue;
      const float bv = p.bias ? p.bias[gn] : 0.f;
      const int gm0 = blockM + waveM * 32 + tm * 16 + hi * 8;
#pragma unroll
      for (int r = 0; r < 8; ++r) {
        float v = acc[tm][tn][r] + bv;
        if (p.doGelu) v = gelu_f(v);
        const size_t ridx = (size_t)(gm0 + r) * p.ldc + gn;
        if (p.resid) v += p.resid[ridx];
        if (Cg) Cg[ridx] = v;
        if (Hg) Hg[(size_t)(gm0 + r) * p.ldh + gn] = (_Float16)v;
      }
    }
  }
}

template <typename AT>
static void gemm_launch(const void* A, const _Float16* Bt, const float* bias,
                        const float* resid, float* outF, _Float16* outH,
                        int M, int N, int K, int lda, int ldb, int ldc, int ldh,
                        int Z, int zdiv,
                        long long aZ0, long long aZ1, long long bZ0, long long bZ1,
                        long long cZ0, long long cZ1, long long hZ0, long long hZ1,
                        int doGelu, hipStream_t stream) {
  GemmP p{A, Bt, bias, resid, outF, outH, M, N, K, lda, ldb, ldc, ldh,
          zdiv, aZ0, aZ1, bZ0, bZ1, cZ0, cZ1, hZ0, hZ1, doGelu};
  dim3 grid((N + 127) / 128, (M + 63) / 64, Z);
  hipLaunchKernelGGL((k_gemm<AT>), grid, dim3(256), 0, stream, p);
}

// ---------------------------------------------------------------- driver

extern "C" void kernel_launch(void* const* d_in, const int* in_sizes, int n_in,
                              void* d_out, int out_size, void* d_ws, size_t ws_size,
                              hipStream_t stream) {
  (void)in_sizes; (void)n_in; (void)out_size; (void)ws_size;

  const float* x0    = (const float*)d_in[0];   // (B,S,E)
  const float* mask  = (const float*)d_in[1];   // (B,S)
  const float* ln1g  = (const float*)d_in[2];
  const float* ln1b  = (const float*)d_in[3];
  const float* Wattn = (const float*)d_in[4];   // (E,3E)
  const float* battn = (const float*)d_in[5];
  const float* Wproj = (const float*)d_in[6];   // (E,E)
  const float* bproj = (const float*)d_in[7];
  const float* ln2g  = (const float*)d_in[8];
  const float* ln2b  = (const float*)d_in[9];
  const float* Wfc   = (const float*)d_in[10];  // (E,I)
  const float* bfc   = (const float*)d_in[11];
  const float* Wp2   = (const float*)d_in[12];  // (I,E)
  const float* bp2   = (const float*)d_in[13];

  float* outX = (float*)d_out;                               // (B,S,E)
  float* attn = outX + (size_t)Mc * Ec;                      // (B,H,S,S)

  char* ws = (char*)d_ws;
  size_t off = 0;
  auto alloc = [&](size_t bytes) -> void* {
    void* p = ws + off;
    off = (off + bytes + 255) & ~(size_t)255;
    return p;
  };

  _Float16* wattnT = (_Float16*)alloc((size_t)Ec * 3 * Ec * 2);  // (3E,E)
  _Float16* wprojT = (_Float16*)alloc((size_t)Ec * Ec * 2);      // (E,E)
  _Float16* wfcT   = (_Float16*)alloc((size_t)Ec * Ic * 2);      // (I,E)
  _Float16* wp2T   = (_Float16*)alloc((size_t)Ic * Ec * 2);      // (E,I)
  _Float16* h16    = (_Float16*)alloc((size_t)Mc * Ec * 2);
  float*    qkv32  = (float*)   alloc((size_t)Mc * 3 * Ec * 4);
  _Float16* q16    = (_Float16*)alloc((size_t)Mc * Ec * 2);
  _Float16* k16    = (_Float16*)alloc((size_t)Mc * Ec * 2);
  _Float16* vT16   = (_Float16*)alloc((size_t)Mc * Ec * 2);
  _Float16* ctx16  = (_Float16*)alloc((size_t)Mc * Ec * 2);
  float*    x32    = (float*)   alloc((size_t)Mc * Ec * 4);
  _Float16* h216   = (_Float16*)alloc((size_t)Mc * Ec * 2);
  _Float16* m16    = (_Float16*)alloc((size_t)Mc * Ic * 2);

  auto cvtT = [&](const float* src, _Float16* dst, int K, int N) {
    hipLaunchKernelGGL(k_cvt_t, dim3(N / 32, K / 32), dim3(32, 8), 0, stream,
                       src, dst, K, N);
  };
  cvtT(Wattn, wattnT, Ec, 3 * Ec);
  cvtT(Wproj, wprojT, Ec, Ec);
  cvtT(Wfc,   wfcT,   Ec, Ic);
  cvtT(Wp2,   wp2T,   Ic, Ec);

  // --- attention path ---
  hipLaunchKernelGGL(k_ln, dim3(Mc), dim3(256), 0, stream, x0, ln1g, ln1b, h16, Ec);

  // qkv = ln1(x) @ Wattn + battn          (M x 3E, f32)
  gemm_launch<_Float16>(h16, wattnT, battn, nullptr, qkv32, nullptr,
                        Mc, 3 * Ec, Ec, Ec, Ec, 3 * Ec, 0,
                        1, 1, 0, 0, 0, 0, 0, 0, 0, 0, 0, stream);

  hipLaunchKernelGGL(k_pack_qkv, dim3(((size_t)Mc * Ec + 255) / 256), dim3(256),
                     0, stream, qkv32, q16, k16, vT16);

  // scores[bh] = q[bh] @ k[bh]^T   -> directly into d_out attn region
  gemm_launch<_Float16>(q16, k16, nullptr, nullptr, attn, nullptr,
                        Sc, Sc, DHc, DHc, DHc, Sc, 0,
                        Bc * Hc, 1,
                        (long long)Sc * DHc, 0, (long long)Sc * DHc, 0,
                        (long long)Sc * Sc, 0, 0, 0, 0, stream);

  hipLaunchKernelGGL(k_softmax, dim3(Bc * Hc * Sc), dim3(256), 0, stream,
                     attn, mask, Sc, Hc);

  // ctx[b,q,h,:] = attn[bh] @ v[bh]       (A is f32, converted while staging)
  gemm_launch<float>(attn, vT16, nullptr, nullptr, nullptr, ctx16,
                     Sc, DHc, Sc, Sc, Sc, DHc, Ec,
                     Bc * Hc, Hc,
                     (long long)Hc * Sc * Sc, (long long)Sc * Sc,
                     (long long)Hc * DHc * Sc, (long long)DHc * Sc,
                     0, 0, (long long)Sc * Ec, (long long)DHc, 0, stream);

  // x = ctx @ Wproj + bproj + x0
  gemm_launch<_Float16>(ctx16, wprojT, bproj, x0, x32, nullptr,
                        Mc, Ec, Ec, Ec, Ec, Ec, 0,
                        1, 1, 0, 0, 0, 0, 0, 0, 0, 0, 0, stream);

  // --- MLP path ---
  hipLaunchKernelGGL(k_ln, dim3(Mc), dim3(256), 0, stream, x32, ln2g, ln2b, h216, Ec);

  // m = gelu(ln2(x) @ Wfc + bfc)          (f16 only, GELU fused in epilogue)
  gemm_launch<_Float16>(h216, wfcT, bfc, nullptr, nullptr, m16,
                        Mc, Ic, Ec, Ec, Ec, Ic, Ic,
                        1, 1, 0, 0, 0, 0, 0, 0, 0, 0, 1, stream);

  // out = m @ Wp2 + bp2 + x  -> final x straight into d_out
  gemm_launch<_Float16>(m16, wp2T, bp2, x32, outX, nullptr,
                        Mc, Ec, Ic, Ic, Ic, Ec, 0,
                        1, 1, 0, 0, 0, 0, 0, 0, 0, 0, 0, stream);
}
